// BoxMaskIoUMetric_40398462386226
// MI455X (gfx1250) — matrix operand order repaired and checked by reading, not compiled, not run on hardware
//
#include <hip/hip_runtime.h>

// CDNA5 (gfx1250) wave32 WMMA types
typedef __attribute__((ext_vector_type(16))) _Float16 v16h;
typedef __attribute__((ext_vector_type(8)))  float    v8f;

// ---------------------------------------------------------------------------
// BoxMaskIoU on MI455X:
// per-sample mask rasterization == (512x32)x(32x512) 0/1 GEMM -> count>0.
// K=32 maps exactly onto one V_WMMA_F32_16X16X32_F16 per 16x16 pixel tile.
// Coverage is encoded as 32-bit "which boxes cover this row/col" bitmasks in
// LDS (8KB); fragments are synthesized branch-free from one mask word each.
// ---------------------------------------------------------------------------

#define MAX_S 512

__global__ void boxiou_init_acc(unsigned int* acc) {
    acc[0] = 0u;   // intersection pixel count
    acc[1] = 0u;   // union pixel count
}

// Build an A-matrix fragment (16x32 f16) from a row coverage mask word.
// ISA 7.12.2 layout: lane<16 holds K={0..7,16..23}; lane>=16 holds K={8..15,24..31}.
__device__ __forceinline__ v16h mkfragA(unsigned w, int baseA) {
    const unsigned t = w >> baseA;         // baseA in {0, 8}
    v16h f;
#pragma unroll
    for (int i = 0; i < 16; ++i) {
        const int sh = (i < 8) ? i : (i + 8);   // bits 0..7 then 16..23
        f[i] = ((t >> sh) & 1u) ? (_Float16)1.0f : (_Float16)0.0f;
    }
    return f;
}

// Build a B-matrix fragment (32x16 f16) from a column coverage mask word.
// Layout: lane<16 holds K={0..7,16..23}; lane>=16 holds K={8..15,24..31}.
__device__ __forceinline__ v16h mkfragB(unsigned w, int baseB) {
    const unsigned t = w >> baseB;         // baseB in {0, 8}
    v16h f;
#pragma unroll
    for (int i = 0; i < 16; ++i)
        f[i] = ((t >> ((i < 8) ? i : (i + 8))) & 1u) ? (_Float16)1.0f : (_Float16)0.0f;
    return f;
}

__global__ __launch_bounds__(256)
void boxiou_main(const float* __restrict__ pred,   // [N][32][6] cx,cy,w,h,cls,obj
                 const float* __restrict__ tgt,    // [N][32][5] cx,cy,w,h,cls
                 const int*   __restrict__ img_size_p,
                 unsigned int* __restrict__ acc)
{
    __shared__ int s_pxlo[32], s_pxhi[32], s_pylo[32], s_pyhi[32];
    __shared__ int s_txlo[32], s_txhi[32], s_tylo[32], s_tyhi[32];
    __shared__ unsigned s_rm_p[MAX_S], s_rm_t[MAX_S];   // row masks (pred/tgt)
    __shared__ unsigned s_cm_p[MAX_S], s_cm_t[MAX_S];   // col masks (pred/tgt)
    __shared__ unsigned int s_inter, s_uni;

    const int n    = blockIdx.x;
    const int tid  = threadIdx.x;
    const int lane = tid & 31;   // wave32
    const int wv   = tid >> 5;   // 8 waves / block
    const int S    = *img_size_p;          // 512
    const float Sf = (float)S;

    if (tid == 0) { s_inter = 0u; s_uni = 0u; }

    // --- phase 1: per-box integer bounds (torch: trunc to int, clamp_max(S)) ---
    if (tid < 32) {
        const float* b = pred + (size_t)n * 32 * 6 + (size_t)tid * 6;
        const float cx = b[0] * Sf, cy = b[1] * Sf;
        const float w  = b[2] * Sf, h  = b[3] * Sf;
        const bool valid = b[5] > 0.5f;    // obj threshold
        int x1 = min((int)(cx - 0.5f * w), S);
        int x2 = min((int)(cx + 0.5f * w), S);
        int y1 = min((int)(cy - 0.5f * h), S);
        int y2 = min((int)(cy + 0.5f * h), S);
        // invalid box -> empty interval (matches "& valid" on ym and xm)
        s_pxlo[tid] = valid ? x1 : 0;  s_pxhi[tid] = valid ? x2 : 0;
        s_pylo[tid] = valid ? y1 : 0;  s_pyhi[tid] = valid ? y2 : 0;
    } else if (tid < 64) {
        const int m = tid - 32;
        const float* b = tgt + (size_t)n * 32 * 5 + (size_t)m * 5;
        const float cx = b[0] * Sf, cy = b[1] * Sf;
        const float w  = b[2] * Sf, h  = b[3] * Sf;
        s_txlo[m] = min((int)(cx - 0.5f * w), S);
        s_txhi[m] = min((int)(cx + 0.5f * w), S);
        s_tylo[m] = min((int)(cy - 0.5f * h), S);
        s_tyhi[m] = min((int)(cy + 0.5f * h), S);
    }
    __syncthreads();

    // --- phase 2: 32-bit coverage bitmasks per pixel row / column ---
    for (int pos = tid; pos < S; pos += 256) {
        unsigned mrp = 0u, mrt = 0u, mcp = 0u, mct = 0u;
#pragma unroll
        for (int m = 0; m < 32; ++m) {
            // (unsigned)(pos-lo) < (unsigned)(hi-lo): single-compare interval test
            mrp |= ((unsigned)(pos - s_pylo[m]) < (unsigned)(s_pyhi[m] - s_pylo[m]) ? 1u : 0u) << m;
            mrt |= ((unsigned)(pos - s_tylo[m]) < (unsigned)(s_tyhi[m] - s_tylo[m]) ? 1u : 0u) << m;
            mcp |= ((unsigned)(pos - s_pxlo[m]) < (unsigned)(s_pxhi[m] - s_pxlo[m]) ? 1u : 0u) << m;
            mct |= ((unsigned)(pos - s_txlo[m]) < (unsigned)(s_txhi[m] - s_txlo[m]) ? 1u : 0u) << m;
        }
        s_rm_p[pos] = mrp;  s_rm_t[pos] = mrt;
        s_cm_p[pos] = mcp;  s_cm_t[pos] = mct;
    }
    __syncthreads();

    // --- phase 3: WMMA tile sweep (branch-free, EXEC all-1s throughout) ---
    const int baseA = (lane < 16) ? 0 : 8;
    const int baseB = (lane < 16) ? 0 : 8;
    const int mn    = lane & 15;
    const int nt    = S >> 4;              // 32 tile rows/cols

    unsigned int my_inter = 0u, my_uni = 0u;

    for (int tr = wv; tr < nt; tr += 8) {
        const int hrow = tr * 16 + mn;     // this lane's matrix row (pixel y)
        const unsigned wrP = s_rm_p[hrow];
        const unsigned wrT = s_rm_t[hrow];
        const v16h aP = mkfragA(wrP, baseA);
        const v16h aT = mkfragA(wrT, baseA);

        for (int tc = 0; tc < nt; ++tc) {
            const int wcol = tc * 16 + mn; // this lane's matrix col (pixel x)
            const v16h bP = mkfragB(s_cm_p[wcol], baseB);
            const v16h bT = mkfragB(s_cm_t[wcol], baseB);

            v8f cz = {};
            // D = A x B (+0): per-pixel box-cover counts; exact in f16/f32 (<=32)
            v8f dP = __builtin_amdgcn_wmma_f32_16x16x32_f16(
                false, aP, false, bP, (short)0, cz, false, false);
            v8f dT = __builtin_amdgcn_wmma_f32_16x16x32_f16(
                false, aT, false, bT, (short)0, cz, false, false);

#pragma unroll
            for (int r = 0; r < 8; ++r) {
                my_inter += (fminf(dP[r], dT[r]) > 0.5f) ? 1u : 0u;
                my_uni   += (fmaxf(dP[r], dT[r]) > 0.5f) ? 1u : 0u;
            }
        }
    }

    // block reduction in LDS, then one global atomic per block
    atomicAdd(&s_inter, my_inter);
    atomicAdd(&s_uni,   my_uni);
    __syncthreads();
    if (tid == 0) {
        atomicAdd(&acc[0], s_inter);
        atomicAdd(&acc[1], s_uni);
    }
}

__global__ void boxiou_finalize(const unsigned int* __restrict__ acc,
                                float* __restrict__ out)
{
    const float inter = (float)acc[0];
    const float uni   = (float)acc[1];
    out[0] = inter / fmaxf(uni, 1.0f);
}

extern "C" void kernel_launch(void* const* d_in, const int* in_sizes, int n_in,
                              void* d_out, int out_size, void* d_ws, size_t ws_size,
                              hipStream_t stream) {
    const float* pred = (const float*)d_in[0];   // [N,32,6] f32
    const float* tgt  = (const float*)d_in[1];   // [N,32,5] f32
    const int*   img  = (const int*)d_in[2];     // scalar img_size
    unsigned int* acc = (unsigned int*)d_ws;     // 2 x u32 accumulators
    float* out = (float*)d_out;                  // scalar IoU (f32)

    const int N = in_sizes[0] / (32 * 6);        // 128

    boxiou_init_acc<<<1, 1, 0, stream>>>(acc);
    boxiou_main<<<N, 256, 0, stream>>>(pred, tgt, img, acc);
    boxiou_finalize<<<1, 1, 0, stream>>>(acc, out);
}
